// PlasticityModelMoE_64209761075260
// MI455X (gfx1250) — matrix operand order repaired and checked by reading, not compiled, not run on hardware
//
#include <hip/hip_runtime.h>
#include <hip/hip_bf16.h>

// ---------------------------------------------------------------------------
// Types for CDNA5 WMMA
// ---------------------------------------------------------------------------
typedef __attribute__((ext_vector_type(16))) __bf16 v16bf;
typedef __attribute__((ext_vector_type(8)))  float  v8f;

union Frag16 { uint4 q[2]; v16bf v; };

__device__ __forceinline__ unsigned short f2bf(float f) {
    unsigned int u = __float_as_uint(f);
    unsigned int r = u + 0x7FFFu + ((u >> 16) & 1u);
    return (unsigned short)(r >> 16);
}

// CDNA5 async global->LDS copy (16B per lane), tracked by ASYNCcnt.
// Takes the real LDS pointer: the ptrtoint both yields the wave-relative LDS
// byte offset (low 32 bits of the shared-aperture address) for the VDST VGPR
// and CAPTURES the LDS object, so the "memory" clobber aliases it and the
// compiler cannot fold the subsequent ds_load fragments to undef.
__device__ __forceinline__ void async_load_b128(const unsigned short* lds_ptr,
                                                const void* g) {
    const unsigned lds_off = (unsigned)(size_t)lds_ptr;
    asm volatile("global_load_async_to_lds_b128 %0, %1, off"
                 :: "v"(lds_off), "v"(g) : "memory");
}
__device__ __forceinline__ void wait_async0() {
    asm volatile("s_wait_asynccnt 0x0" ::: "memory");
}

// ---------------------------------------------------------------------------
// Generic bf16 WMMA GEMM: C(f32, MxN) = A(bf16, MxK, row-major) @ B
// with B supplied transposed: Bt(bf16, NxK, row-major).
// Requires M%256==0, N%128==0, K%32==0.
// Block: 256 threads (8 waves, 4x2 wave grid), block tile 256x128,
// wave tile 64x64 (4x4 WMMA tiles), K-step 32, double-buffered LDS filled
// by global_load_async_to_lds_b128 one K-step ahead.
// ---------------------------------------------------------------------------
#define LDSSTR 40                         // 32 + 8 pad (80B rows, 16B aligned)
#define AROWS  256
#define BROWS  128
#define BUFELEMS ((AROWS + BROWS) * LDSSTR)   // 15360 bf16 per stage

__global__ __launch_bounds__(256)
void k_gemm_bf16(const unsigned short* __restrict__ A,
                 const unsigned short* __restrict__ Bt,
                 float* __restrict__ C, int M, int N, int K) {
    __shared__ __align__(16) unsigned short SL[2 * BUFELEMS];  // 61440 B

    const int t    = threadIdx.x;
    const int lane = t & 31;
    const int wave = t >> 5;
    const int wr   = wave >> 1;   // 0..3 -> 64-row band
    const int wc   = wave & 1;    // 0..1 -> 64-col band
    const int m0   = blockIdx.y * AROWS;
    const int n0   = blockIdx.x * BROWS;

    const int lrow  = lane & 15;
    const int ahalf = (lane >> 4) * 8;    // A frag K-offset: 0 or 8
    const int bhalf = (lane >> 4) * 8;    // B frag K-offset: 0 or 8

    v8f acc[4][4];
#pragma unroll
    for (int mi = 0; mi < 4; ++mi)
#pragma unroll
        for (int ni = 0; ni < 4; ++ni)
            acc[mi][ni] = (v8f){0.f, 0.f, 0.f, 0.f, 0.f, 0.f, 0.f, 0.f};

    // Stage one 256x32 A tile + 128x32 Bt tile into LDS buffer `buf` (async).
    auto stage = [&](int buf, int kb) {
        const unsigned base = (unsigned)buf * BUFELEMS;
#pragma unroll
        for (int j = 0; j < 4; ++j) {                 // 1024 A chunks
            const int c = t + j * 256;
            const int row = c >> 2, quad = c & 3;
            async_load_b128(&SL[base + row * LDSSTR + quad * 8],
                            A + (size_t)(m0 + row) * K + kb + quad * 8);
        }
#pragma unroll
        for (int j = 0; j < 2; ++j) {                 // 512 B chunks
            const int c = t + j * 256;
            const int row = c >> 2, quad = c & 3;
            async_load_b128(&SL[base + AROWS * LDSSTR + row * LDSSTR + quad * 8],
                            Bt + (size_t)(n0 + row) * K + kb + quad * 8);
        }
    };

    const int nk = K / 32;
    stage(0, 0);
    for (int i = 0; i < nk; ++i) {
        wait_async0();        // this wave's fills for buf (i&1) have landed
        __syncthreads();      // everyone's fills landed; prev reads drained
        if (i + 1 < nk) stage((i + 1) & 1, (i + 1) * 32);

        const unsigned base = (unsigned)(i & 1) * BUFELEMS;
        Frag16 afr[4], bfr[4];
#pragma unroll
        for (int mi = 0; mi < 4; ++mi) {
            const unsigned off = base + (wr * 64 + mi * 16 + lrow) * LDSSTR + ahalf;
            afr[mi].q[0] = *reinterpret_cast<const uint4*>(&SL[off]);
            afr[mi].q[1] = *reinterpret_cast<const uint4*>(&SL[off + 16]);
        }
#pragma unroll
        for (int ni = 0; ni < 4; ++ni) {
            const unsigned off = base + AROWS * LDSSTR +
                                 (wc * 64 + ni * 16 + lrow) * LDSSTR + bhalf;
            bfr[ni].q[0] = *reinterpret_cast<const uint4*>(&SL[off]);
            bfr[ni].q[1] = *reinterpret_cast<const uint4*>(&SL[off + 16]);
        }
#pragma unroll
        for (int mi = 0; mi < 4; ++mi)
#pragma unroll
            for (int ni = 0; ni < 4; ++ni)
                acc[mi][ni] = __builtin_amdgcn_wmma_f32_16x16x32_bf16(
                    false, afr[mi].v, false, bfr[ni].v,
                    (short)0, acc[mi][ni], false, false);
    }

    // Epilogue: f32 C/D layout — VGPR r: row = r (+8 for lanes>=16), col = lane&15.
    const int rowoff = (lane >> 4) * 8;
    const int coll   = lane & 15;
#pragma unroll
    for (int mi = 0; mi < 4; ++mi)
#pragma unroll
        for (int ni = 0; ni < 4; ++ni)
#pragma unroll
            for (int r = 0; r < 8; ++r) {
                const int gr = m0 + wr * 64 + mi * 16 + r + rowoff;
                const int gc = n0 + wc * 64 + ni * 16 + coll;
                C[(size_t)gr * N + gc] = acc[mi][ni][r];
            }
}

// ---------------------------------------------------------------------------
// f32 -> bf16 straight conversion
// ---------------------------------------------------------------------------
__global__ void k_convert_bf16(const float* __restrict__ in,
                               unsigned short* __restrict__ out, int n) {
    int i = blockIdx.x * 256 + threadIdx.x;
    if (i < n) out[i] = f2bf(in[i]);
}

// f32 (R x C) -> bf16 transposed (C x R), LDS-tiled for coalescing. R,C % 32 == 0.
__global__ void k_transpose_bf16(const float* __restrict__ in,
                                 unsigned short* __restrict__ out, int R, int C) {
    __shared__ float tile[32][33];
    const int c0 = blockIdx.x * 32, r0 = blockIdx.y * 32;
    const int tx = threadIdx.x, ty = threadIdx.y;  // block (32, 8)
#pragma unroll
    for (int i = 0; i < 32; i += 8)
        tile[ty + i][tx] = in[(size_t)(r0 + ty + i) * C + c0 + tx];
    __syncthreads();
#pragma unroll
    for (int i = 0; i < 32; i += 8)
        out[(size_t)(c0 + ty + i) * R + r0 + tx] = f2bf(tile[tx][ty + i]);
}

// ---------------------------------------------------------------------------
// Branch gate: softmax(x @ gate_w + gate_b) over 4 branches. One block per row.
// ---------------------------------------------------------------------------
__global__ __launch_bounds__(256)
void k_gate(const float* __restrict__ x, const float* __restrict__ gw,
            const float* __restrict__ gb, float* __restrict__ gate, int D) {
    __shared__ float red[256 * 4];
    const int b = blockIdx.x, t = threadIdx.x;
    float acc[4] = {0.f, 0.f, 0.f, 0.f};
    for (int i = t; i < D; i += 256) {
        const float xv = x[(size_t)b * D + i];
#pragma unroll
        for (int n = 0; n < 4; ++n) acc[n] += xv * gw[i * 4 + n];
    }
#pragma unroll
    for (int n = 0; n < 4; ++n) red[t * 4 + n] = acc[n];
    __syncthreads();
    for (int s = 128; s > 0; s >>= 1) {
        if (t < s)
#pragma unroll
            for (int n = 0; n < 4; ++n) red[t * 4 + n] += red[(t + s) * 4 + n];
        __syncthreads();
    }
    if (t == 0) {
        float l[4], m = -3.4e38f, s = 0.f;
#pragma unroll
        for (int n = 0; n < 4; ++n) { l[n] = red[n] + gb[n]; m = fmaxf(m, l[n]); }
#pragma unroll
        for (int n = 0; n < 4; ++n) { l[n] = expf(l[n] - m); s += l[n]; }
#pragma unroll
        for (int n = 0; n < 4; ++n) gate[b * 4 + n] = l[n] / s;
    }
}

// softmax of the 9 activation weights (tiny, once)
__global__ void k_w9(const float* __restrict__ aw, float* __restrict__ w9) {
    if (threadIdx.x == 0 && blockIdx.x == 0) {
        float m = -3.4e38f;
        for (int i = 0; i < 9; ++i) m = fmaxf(m, aw[i]);
        float e[9], s = 0.f;
        for (int i = 0; i < 9; ++i) { e[i] = expf(aw[i] - m); s += e[i]; }
        for (int i = 0; i < 9; ++i) w9[i] = e[i] / s;
    }
}

__device__ __forceinline__ float softplusf(float x) {
    return x > 20.f ? x : log1pf(expf(x));
}

// hidden = quantum_act(sum_n (pre+b)[b,u,n]*gate[b,n]) * mask[u]; also bf16 copy
__global__ __launch_bounds__(256)
void k_hidden(const float* __restrict__ pre, const float* __restrict__ bias,
              const float* __restrict__ gate, const float* __restrict__ w9,
              const float* __restrict__ mask, float* __restrict__ hidden,
              unsigned short* __restrict__ hbf, int U) {
    const int idx = blockIdx.x * 256 + threadIdx.x;
    const int b = idx / U, u = idx - b * U;
    const float4 p  = *reinterpret_cast<const float4*>(&pre[(size_t)idx * 4]);
    const float4 bi = *reinterpret_cast<const float4*>(&bias[u * 4]);
    const float4 g  = *reinterpret_cast<const float4*>(&gate[b * 4]);
    const float s = (p.x + bi.x) * g.x + (p.y + bi.y) * g.y +
                    (p.z + bi.z) * g.z + (p.w + bi.w) * g.w;
    const float sig  = 1.f / (1.f + expf(-s));
    const float elu  = s > 0.f ? s : expf(s) - 1.f;
    const float th   = tanhf(s);
    const float rl   = fmaxf(s, 0.f);
    const float silu = s * sig;
    const float gelu = 0.5f * s * (1.f + erff(s * 0.70710678118654752f));
    const float selu = 1.0507009873554805f *
                       (s > 0.f ? s : 1.6732632423543772f * (expf(s) - 1.f));
    const float mish = s * tanhf(softplusf(s));
    const float h = (w9[0] * sig + w9[1] * elu + w9[2] * th + w9[3] * rl +
                     w9[4] * silu + w9[5] * gelu + w9[6] * selu + w9[7] * mish +
                     w9[8] * s) * mask[u];
    hidden[idx] = h;
    hbf[idx] = f2bf(h);
}

// Row softmax with bias (in place) + bf16 copy. One block per row.
__global__ __launch_bounds__(256)
void k_softmax_rows(float* __restrict__ a, const float* __restrict__ bias,
                    unsigned short* __restrict__ obf, int N) {
    __shared__ float red[256];
    const int b = blockIdx.x, t = threadIdx.x;
    float m = -3.4e38f;
    for (int i = t; i < N; i += 256) {
        const float v = a[(size_t)b * N + i] + bias[i];
        a[(size_t)b * N + i] = v;
        m = fmaxf(m, v);
    }
    red[t] = m; __syncthreads();
    for (int s = 128; s > 0; s >>= 1) {
        if (t < s) red[t] = fmaxf(red[t], red[t + s]);
        __syncthreads();
    }
    const float mx = red[0]; __syncthreads();
    float sum = 0.f;
    for (int i = t; i < N; i += 256) {
        const float e = expf(a[(size_t)b * N + i] - mx);
        a[(size_t)b * N + i] = e;
        sum += e;
    }
    red[t] = sum; __syncthreads();
    for (int s = 128; s > 0; s >>= 1) {
        if (t < s) red[t] += red[t + s];
        __syncthreads();
    }
    const float inv = 1.f / red[0];
    for (int i = t; i < N; i += 256) {
        const float v = a[(size_t)b * N + i] * inv;
        a[(size_t)b * N + i] = v;
        obf[(size_t)b * N + i] = f2bf(v);
    }
}

// column mean over B rows
__global__ void k_colmean(const float* __restrict__ in, float* __restrict__ out,
                          int B, int N) {
    const int m = blockIdx.x * 256 + threadIdx.x;
    if (m >= N) return;
    float s = 0.f;
    for (int b = 0; b < B; ++b) s += in[(size_t)b * N + m];
    out[m] = s / (float)B;
}

// column mean of tanh(in + bias)
__global__ void k_colmean_tanh(const float* __restrict__ in,
                               const float* __restrict__ bias,
                               float* __restrict__ out, int B, int N) {
    const int d = blockIdx.x * 256 + threadIdx.x;
    if (d >= N) return;
    float s = 0.f;
    for (int b = 0; b < B; ++b) s += tanhf(in[(size_t)b * N + d] + bias[d]);
    out[d] = s / (float)B;
}

__global__ void k_addbias(float* __restrict__ a, const float* __restrict__ bias,
                          int n, int O) {
    const int i = blockIdx.x * 256 + threadIdx.x;
    if (i < n) a[i] += bias[i % O];
}

// new_mem = mem*(1-u) + u*cand_mean;  u[m] = colmean_att[m]*update_gate[m]
__global__ void k_newmem(const float* __restrict__ mem,
                         const float* __restrict__ cm_att,
                         const float* __restrict__ ug,
                         const float* __restrict__ cand_mean,
                         float* __restrict__ out, int MD, int n) {
    const int i = blockIdx.x * 256 + threadIdx.x;
    if (i >= n) return;
    const int m = i / MD, d = i - m * MD;
    const float u = cm_att[m] * ug[m];
    out[i] = mem[i] * (1.f - u) + u * cand_mean[d];
}

// Critic + gated prediction output. One block per batch row.
__global__ __launch_bounds__(256)
void k_critic(const float* __restrict__ read, const float* __restrict__ preds,
              const float* __restrict__ hidden,
              const float* __restrict__ cw1, const float* __restrict__ cb1,
              const float* __restrict__ cw2, const float* __restrict__ cb2,
              const float* __restrict__ cw3, const float* __restrict__ cb3,
              const float* __restrict__ cw4, const float* __restrict__ cb4,
              const float* __restrict__ cw5, const float* __restrict__ cb5,
              const float* __restrict__ thr, float* __restrict__ out) {
    __shared__ float sread[1024];
    __shared__ float spred[512];
    __shared__ float shid[2048];
    __shared__ float feat[96];
    __shared__ float fused[64];
    __shared__ float sgate;
    const int b = blockIdx.x, t = threadIdx.x;
    for (int i = t; i < 1024; i += 256) sread[i] = read[(size_t)b * 1024 + i];
    for (int i = t; i < 512;  i += 256) spred[i] = preds[(size_t)b * 512 + i];
    for (int i = t; i < 2048; i += 256) shid[i]  = hidden[(size_t)b * 2048 + i];
    __syncthreads();
    if (t < 96) {
        const int j = t & 31;
        const float* src; const float* W; float acc; int K;
        if (t < 32)      { src = sread; W = cw1; acc = cb1[j]; K = 1024; }
        else if (t < 64) { src = spred; W = cw2; acc = cb2[j]; K = 512;  }
        else             { src = shid;  W = cw3; acc = cb3[j]; K = 2048; }
        for (int k = 0; k < K; ++k) acc += src[k] * W[k * 32 + j];
        feat[t] = fmaxf(acc, 0.f);
    }
    __syncthreads();
    if (t < 64) {
        float acc = cb4[t];
        for (int k = 0; k < 96; ++k) acc += feat[k] * cw4[k * 64 + t];
        fused[t] = fmaxf(acc, 0.f);
    }
    __syncthreads();
    if (t == 0) {
        float o0 = cb5[0];
        for (int k = 0; k < 64; ++k) o0 += fused[k] * cw5[k * 2];
        sgate = 1.f / (1.f + expf(-(o0 - thr[0])));
    }
    __syncthreads();
    for (int o = t; o < 512; o += 256) out[(size_t)b * 512 + o] = spred[o] * sgate;
}

// ---------------------------------------------------------------------------
// Host launch
// ---------------------------------------------------------------------------
extern "C" void kernel_launch(void* const* d_in, const int* in_sizes, int n_in,
                              void* d_out, int out_size, void* d_ws, size_t ws_size,
                              hipStream_t stream) {
    (void)in_sizes; (void)n_in; (void)out_size; (void)ws_size;
    constexpr int B = 1024, D = 1024, U = 2048, NBR = 4;
    constexpr int M = 2048, MD = 1024, O = 512;
    constexpr int UN = U * NBR;  // 8192

    const float* x        = (const float*)d_in[0];
    const float* w        = (const float*)d_in[1];   // (D, U, N) == row-major D x UN
    const float* bb       = (const float*)d_in[2];
    const float* gate_w   = (const float*)d_in[3];
    const float* gate_b   = (const float*)d_in[4];
    const float* act_w    = (const float*)d_in[5];
    const float* mask     = (const float*)d_in[6];
    const float* mem      = (const float*)d_in[7];
    const float* att_w    = (const float*)d_in[8];
    const float* att_b    = (const float*)d_in[9];
    const float* write_w  = (const float*)d_in[10];
    const float* write_b  = (const float*)d_in[11];
    const float* ug       = (const float*)d_in[12];
    const float* out_w    = (const float*)d_in[13];
    const float* out_b    = (const float*)d_in[14];
    const float* cw1 = (const float*)d_in[15]; const float* cb1 = (const float*)d_in[16];
    const float* cw2 = (const float*)d_in[17]; const float* cb2 = (const float*)d_in[18];
    const float* cw3 = (const float*)d_in[19]; const float* cb3 = (const float*)d_in[20];
    const float* cw4 = (const float*)d_in[21]; const float* cb4 = (const float*)d_in[22];
    const float* cw5 = (const float*)d_in[23]; const float* cb5 = (const float*)d_in[24];
    const float* thr = (const float*)d_in[25];

    char* ws = (char*)d_ws;
    size_t off = 0;
    auto alloc = [&](size_t bytes) -> void* {
        void* p = ws + off;
        off = (off + bytes + 255) & ~(size_t)255;
        return p;
    };

    unsigned short* xbf   = (unsigned short*)alloc((size_t)B * D * 2);
    unsigned short* w_t   = (unsigned short*)alloc((size_t)UN * D * 2);
    unsigned short* aw_t  = (unsigned short*)alloc((size_t)M * U * 2);
    unsigned short* ww_t  = (unsigned short*)alloc((size_t)MD * U * 2);
    unsigned short* ow_t  = (unsigned short*)alloc((size_t)O * U * 2);
    unsigned short* mem_t = (unsigned short*)alloc((size_t)MD * M * 2);
    unsigned short* hbf   = (unsigned short*)alloc((size_t)B * U * 2);
    unsigned short* rabf  = (unsigned short*)alloc((size_t)B * M * 2);
    float* pre      = (float*)alloc((size_t)B * UN * 4);
    float* gate     = (float*)alloc((size_t)B * NBR * 4);
    float* w9       = (float*)alloc(9 * 4);
    float* hidden   = (float*)alloc((size_t)B * U * 4);
    float* att      = (float*)alloc((size_t)B * M * 4);
    float* readbuf  = (float*)alloc((size_t)B * MD * 4);
    float* candpre  = (float*)alloc((size_t)B * MD * 4);
    float* preds    = (float*)alloc((size_t)B * O * 4);
    float* cm_att   = (float*)alloc((size_t)M * 4);
    float* cm_cand  = (float*)alloc((size_t)MD * 4);

    float* out_preds = (float*)d_out;
    float* out_mem   = (float*)d_out + (size_t)B * O;

    const dim3 blk256(256);
    const dim3 tblk(32, 8);

    // --- precision conversions / transposes (one-time per launch) ---
    k_convert_bf16<<<dim3((B * D) / 256), blk256, 0, stream>>>(x, xbf, B * D);
    k_transpose_bf16<<<dim3(UN / 32, D / 32), tblk, 0, stream>>>(w, w_t, D, UN);
    k_transpose_bf16<<<dim3(M / 32, U / 32),  tblk, 0, stream>>>(att_w, aw_t, U, M);
    k_transpose_bf16<<<dim3(MD / 32, U / 32), tblk, 0, stream>>>(write_w, ww_t, U, MD);
    k_transpose_bf16<<<dim3(O / 32, U / 32),  tblk, 0, stream>>>(out_w, ow_t, U, O);
    k_transpose_bf16<<<dim3(MD / 32, M / 32), tblk, 0, stream>>>(mem, mem_t, M, MD);

    // --- branch gate + activation-mix weights ---
    k_gate<<<dim3(B), blk256, 0, stream>>>(x, gate_w, gate_b, gate, D);
    k_w9<<<dim3(1), dim3(32), 0, stream>>>(act_w, w9);

    // --- GEMM 1: pre = x @ w  (1024 x 8192 x 1024) ---
    k_gemm_bf16<<<dim3(UN / 128, B / 256), blk256, 0, stream>>>(xbf, w_t, pre, B, UN, D);
    k_hidden<<<dim3((B * U) / 256), blk256, 0, stream>>>(pre, bb, gate, w9, mask, hidden, hbf, U);

    // --- GEMM 2: attention logits = hidden @ att_w (1024 x 2048 x 2048) ---
    k_gemm_bf16<<<dim3(M / 128, B / 256), blk256, 0, stream>>>(hbf, aw_t, att, B, M, U);
    k_softmax_rows<<<dim3(B), blk256, 0, stream>>>(att, att_b, rabf, M);
    k_colmean<<<dim3(M / 256), blk256, 0, stream>>>(att, cm_att, B, M);

    // --- GEMM 3: read = read_att @ mem (1024 x 1024 x 2048) ---
    k_gemm_bf16<<<dim3(MD / 128, B / 256), blk256, 0, stream>>>(rabf, mem_t, readbuf, B, MD, M);

    // --- GEMM 4: cand_pre = hidden @ write_w (1024 x 1024 x 2048) ---
    k_gemm_bf16<<<dim3(MD / 128, B / 256), blk256, 0, stream>>>(hbf, ww_t, candpre, B, MD, U);
    k_colmean_tanh<<<dim3(MD / 256), blk256, 0, stream>>>(candpre, write_b, cm_cand, B, MD);

    // --- memory update -> second output segment ---
    k_newmem<<<dim3((M * MD) / 256), blk256, 0, stream>>>(mem, cm_att, ug, cm_cand,
                                                          out_mem, MD, M * MD);

    // --- GEMM 5: preds = hidden @ out_w (1024 x 512 x 2048) ---
    k_gemm_bf16<<<dim3(O / 128, B / 256), blk256, 0, stream>>>(hbf, ow_t, preds, B, O, U);
    k_addbias<<<dim3((B * O) / 256), blk256, 0, stream>>>(preds, out_b, B * O, O);

    // --- critic + gated predictions -> first output segment ---
    k_critic<<<dim3(B), blk256, 0, stream>>>(readbuf, preds, hidden,
                                             cw1, cb1, cw2, cb2, cw3, cb3,
                                             cw4, cb4, cw5, cb5, thr, out_preds);
}
